// SelfAttention_15848429322967
// MI455X (gfx1250) — compile-verified, matrix operand order checked by reading, not run on hardware
//
#include <hip/hip_runtime.h>

#define GROUPS 256
#define SEQ    512
#define NTOK   (GROUPS * SEQ)
#define D_IN   128
#define D_QK   64
#define D_V    128
#define NIT    (SEQ / 32)

typedef __attribute__((ext_vector_type(16))) __bf16         v16bf;
typedef __attribute__((ext_vector_type(16))) unsigned short v16u;
typedef __attribute__((ext_vector_type(8)))  unsigned short v8u;
typedef __attribute__((ext_vector_type(8)))  float          v8f;
typedef __attribute__((ext_vector_type(4)))  float          v4f;
typedef __attribute__((ext_vector_type(4)))  unsigned int   v4ui;
typedef __attribute__((ext_vector_type(8)))  int            v8i_t;
typedef __attribute__((ext_vector_type(4)))  int            v4i_t;

__device__ __forceinline__ unsigned short f2bf(float f) {
    return __builtin_bit_cast(unsigned short, (__bf16)f);   // native v_cvt bf16
}
__device__ __forceinline__ v16bf as_bf(v16u u) { return __builtin_bit_cast(v16bf, u); }

// raw LDS byte offset of a __shared__ object (AS3 pointer value)
typedef __attribute__((address_space(3))) unsigned short as3_u16;
__device__ __forceinline__ unsigned lds_off_u16(unsigned short* p) {
    return (unsigned)(unsigned long long)(as3_u16*)p;
}

// ---------------------------------------------------------------------------
// TDM: 2-D tile (width_elems x rows, bf16, row stride in elems) -> LDS,
// packed contiguously.  Descriptor per CDNA5 ISA:
//   group0: count=1 | lds_addr | global_addr[56:0] | type=2
//   group1: data_size=1(2B); tensor dims == tile dims (no OOB clipping);
//           tile_dim0/1; tensor_dim0_stride
// ---------------------------------------------------------------------------
__device__ __forceinline__ void tdm_load_2d(unsigned lds_byte_addr,
                                            const void* gptr,
                                            unsigned width_elems,
                                            unsigned rows,
                                            unsigned row_stride_elems)
{
    unsigned long long ga = (unsigned long long)gptr;
    v4ui g0 = { 1u,                                   // count=1, user mode
                lds_byte_addr,
                (unsigned)ga,
                (unsigned)((ga >> 32) & 0x1FFFFFFull) | 0x80000000u };  // type=2
    v8i_t g1 = { (int)(1u << 16),                                      // data_size=2B
                 (int)((width_elems & 0xFFFFu) << 16),                 // tensor_dim0 lo
                 (int)((width_elems >> 16) | ((rows & 0xFFFFu) << 16)),// td0 hi|td1 lo
                 (int)((rows >> 16) | (width_elems << 16)),            // td1 hi|tile_dim0
                 (int)(rows & 0xFFFFu),                                // tile_dim1 (td2=0)
                 (int)row_stride_elems,                                // stride0[31:0]
                 0, 0 };
    v4i_t z4 = {0, 0, 0, 0};
#if defined(__clang_major__) && (__clang_major__ >= 23)
    v8i_t z8 = {0, 0, 0, 0, 0, 0, 0, 0};
    __builtin_amdgcn_tensor_load_to_lds(g0, g1, z4, z4, z8, 0);
#else
    __builtin_amdgcn_tensor_load_to_lds(g0, g1, z4, z4, 0);
#endif
}

// ---------------------------------------------------------------------------
// Kernel 1: fused QKV projection.  X[N,128]f32 @ (Wq|Wk|Wv)[128,256] + bias.
// Weights staged transposed in LDS as bf16 (B-fragment friendly).  Q,K
// written bf16 row-major [N,64]; V written bf16 transposed per group
// Vt[g][d][l] so the attention kernel's P@V B-fragments are contiguous.
// ---------------------------------------------------------------------------
__global__ __launch_bounds__(256) void qkv_proj_kernel(
    const float* __restrict__ X,
    const float* __restrict__ Wq, const float* __restrict__ bq,
    const float* __restrict__ Wk, const float* __restrict__ bk,
    const float* __restrict__ Wv, const float* __restrict__ bv,
    unsigned short* __restrict__ Qw, unsigned short* __restrict__ Kw,
    unsigned short* __restrict__ Vt)
{
    __shared__ unsigned short Wt[256][128];   // 64 KB: Wt[n][d_in] bf16
    __shared__ float          biasl[256];

    const int tid = threadIdx.x;
    for (int idx = tid; idx < 128 * 64; idx += 256) {   // coalesced staging
        const int d = idx >> 6, n = idx & 63;
        Wt[n][d]      = f2bf(Wq[idx]);
        Wt[64 + n][d] = f2bf(Wk[idx]);
    }
    for (int idx = tid; idx < 128 * 128; idx += 256) {
        const int d = idx >> 7, n = idx & 127;
        Wt[128 + n][d] = f2bf(Wv[idx]);
    }
    biasl[tid] = (tid < 64) ? bq[tid] : (tid < 128 ? bk[tid - 64] : bv[tid - 128]);
    __syncthreads();

    const int lane = tid & 31, wave = tid >> 5;
    const int m  = lane & 15;              // row (A) / column (B,D) within tile
    const int kb = (lane >> 4) * 8;        // K-offset of upper lane half
    const int gwave  = blockIdx.x * 8 + wave;
    const int nwaves = gridDim.x * 8;

    for (int tile = gwave; tile < NTOK / 16; tile += nwaves) {
        const int row0 = tile * 16;
        const float* xrow = X + (size_t)(row0 + m) * D_IN;
        if (tile + nwaves < NTOK / 16)     // global_prefetch_b8 next X tile
            __builtin_prefetch(X + (size_t)(row0 + nwaves * 16 + m) * D_IN, 0, 0);

        // A fragments: 16 tokens x 128 d_in = 4 chunks of K=32
        v16u a[4];
#pragma unroll
        for (int c = 0; c < 4; ++c) {
            const int d0 = c * 32 + kb;
            v4f l0 = *(const v4f*)(xrow + d0);
            v4f l1 = *(const v4f*)(xrow + d0 + 4);
            v4f h0 = *(const v4f*)(xrow + d0 + 16);
            v4f h1 = *(const v4f*)(xrow + d0 + 20);
#pragma unroll
            for (int i = 0; i < 4; ++i) {
                a[c][i]      = f2bf(l0[i]);  a[c][4 + i]  = f2bf(l1[i]);
                a[c][8 + i]  = f2bf(h0[i]);  a[c][12 + i] = f2bf(h1[i]);
            }
        }

        for (int nt = 0; nt < 16; ++nt) {               // 16 output col tiles
            const int n = nt * 16 + m;
            v8f acc = {0.f, 0.f, 0.f, 0.f, 0.f, 0.f, 0.f, 0.f};
#pragma unroll
            for (int c = 0; c < 4; ++c) {
                const int d0 = c * 32 + kb;
                const unsigned short* wr = &Wt[n][0];
                v8u bl = *(const v8u*)(wr + d0);        // ds_load_b128
                v8u bh = *(const v8u*)(wr + d0 + 16);
                v16u bu;
#pragma unroll
                for (int i = 0; i < 8; ++i) { bu[i] = bl[i]; bu[8 + i] = bh[i]; }
                acc = __builtin_amdgcn_wmma_f32_16x16x32_bf16(
                        false, as_bf(a[c]), false, as_bf(bu),
                        (short)0, acc, false, false);
            }
            const float bb    = biasl[n];
            const int   mrow0 = row0 + ((lane >> 4) << 3);
            if (nt < 8) {                               // Q or K, row-major
                unsigned short* dst = (nt < 4) ? Qw : Kw;
                const int col = (nt < 4) ? n : (n - 64);
#pragma unroll
                for (int r = 0; r < 8; ++r)
                    dst[(size_t)(mrow0 + r) * D_QK + col] = f2bf(acc[r] + bb);
            } else {                                    // V transposed per group
                const int d  = n - 128;
                const int g  = row0 / SEQ;              // 512 % 16 == 0
                const int l0 = (row0 % SEQ) + ((lane >> 4) << 3);
                v8u pk;
#pragma unroll
                for (int r = 0; r < 8; ++r) pk[r] = f2bf(acc[r] + bb);
                *(v8u*)(Vt + (size_t)g * D_V * SEQ + (size_t)d * SEQ + l0) = pk;
            }
        }
    }
}

// ---------------------------------------------------------------------------
// Kernel 2: flash attention.  One workgroup = (group, 64-query block),
// one wave = 16 query rows.  K/V tiles staged in LDS by the Tensor Data
// Mover, double-buffered (wave0 issues tile it+1, waits TENSORcnt for tile
// it, then a workgroup barrier publishes the LDS data).  Per 32-key
// iteration: 4 WMMAs (S = Q K^T), online softmax in registers, D->A
// relayout of P through wave-private LDS, 8 WMMAs (O += P V).
// ---------------------------------------------------------------------------
__global__ __launch_bounds__(128) void flash_attn_kernel(
    const unsigned short* __restrict__ Qw,
    const unsigned short* __restrict__ Kw,
    const unsigned short* __restrict__ Vt,
    float* __restrict__ Out)
{
    __shared__ unsigned short Kbuf[2][32 * 64];    //  8 KB: [key][d] bf16
    __shared__ unsigned short Vbuf[2][128 * 32];   // 16 KB: [dim][key] bf16
    __shared__ float          Pst[4][16][32];      //  8 KB per-wave P staging

    const int tid  = threadIdx.x;
    const int lane = tid & 31, wave = tid >> 5;
    const int grp   = blockIdx.x;
    const int qrow0 = blockIdx.y * 64 + wave * 16;      // within group
    const int m  = lane & 15;
    const int kb = (lane >> 4) * 8;
    const float sc = 0.0883883476483184f;               // 1/sqrt(2*64)

    const unsigned short* Kg = Kw + (size_t)grp * SEQ * D_QK;
    const unsigned short* Vg = Vt + (size_t)grp * (size_t)D_V * SEQ;

    // Q A-fragments (16 rows x K=64 -> two chunks), resident for whole loop
    const size_t qbase = ((size_t)grp * SEQ + qrow0 + m) * D_QK;
    v16u qa[2];
#pragma unroll
    for (int c = 0; c < 2; ++c) {
        const int d0 = c * 32 + kb;
        v8u lo = *(const v8u*)(Qw + qbase + d0);
        v8u hi = *(const v8u*)(Qw + qbase + d0 + 16);
#pragma unroll
        for (int i = 0; i < 8; ++i) { qa[c][i] = lo[i]; qa[c][8 + i] = hi[i]; }
    }

    v8f o[8];
#pragma unroll
    for (int f = 0; f < 8; ++f)
        o[f] = (v8f){0.f, 0.f, 0.f, 0.f, 0.f, 0.f, 0.f, 0.f};
    float mrow[8], lrow[8];
#pragma unroll
    for (int r = 0; r < 8; ++r) { mrow[r] = -3.0e38f; lrow[r] = 0.f; }

    // TDM prologue: stage tile 0
    if (wave == 0) {
        tdm_load_2d(lds_off_u16(&Kbuf[0][0]), Kg,      64, 32,  64);
        tdm_load_2d(lds_off_u16(&Vbuf[0][0]), Vg,      32, 128, 512);
    }

    for (int it = 0; it < NIT; ++it) {
        if (wave == 0) {
            if (it + 1 < NIT) {                     // issue next, wait current
                const int b = (it + 1) & 1;
                tdm_load_2d(lds_off_u16(&Kbuf[b][0]),
                            Kg + (size_t)(it + 1) * 32 * D_QK, 64, 32, 64);
                tdm_load_2d(lds_off_u16(&Vbuf[b][0]),
                            Vg + (it + 1) * 32,                32, 128, 512);
                __builtin_amdgcn_s_wait_tensorcnt(2);   // tile 'it' complete
            } else {
                __builtin_amdgcn_s_wait_tensorcnt(0);
            }
        }
        __syncthreads();                            // publish LDS tile
        const unsigned short* Kt = &Kbuf[it & 1][0];
        const unsigned short* Vti = &Vbuf[it & 1][0];

        // S = Q K^T over this 32-key window (two 16-key column tiles)
        v8f s0 = {0.f,0.f,0.f,0.f,0.f,0.f,0.f,0.f};
        v8f s1 = {0.f,0.f,0.f,0.f,0.f,0.f,0.f,0.f};
#pragma unroll
        for (int c = 0; c < 2; ++c) {
            const int d0 = c * 32 + kb;
            const unsigned short* kr0 = Kt + (size_t)m * D_QK + d0;
            const unsigned short* kr1 = Kt + (size_t)(16 + m) * D_QK + d0;
            v8u a0 = *(const v8u*)kr0, b0 = *(const v8u*)(kr0 + 16);
            v8u a1 = *(const v8u*)kr1, b1 = *(const v8u*)(kr1 + 16);
            v16u f0, f1;
#pragma unroll
            for (int i = 0; i < 8; ++i) {
                f0[i] = a0[i]; f0[8 + i] = b0[i];
                f1[i] = a1[i]; f1[8 + i] = b1[i];
            }
            s0 = __builtin_amdgcn_wmma_f32_16x16x32_bf16(
                    false, as_bf(qa[c]), false, as_bf(f0), (short)0, s0, false, false);
            s1 = __builtin_amdgcn_wmma_f32_16x16x32_bf16(
                    false, as_bf(qa[c]), false, as_bf(f1), (short)0, s1, false, false);
        }

        // online softmax; row r lives on this lane's 16-lane half
        float alpha[8];
#pragma unroll
        for (int r = 0; r < 8; ++r) {
            const float x0 = s0[r] * sc, x1 = s1[r] * sc;
            float mr = fmaxf(x0, x1);
#pragma unroll
            for (int off = 1; off < 16; off <<= 1)
                mr = fmaxf(mr, __shfl_xor(mr, off, 32));
            const float mn = fmaxf(mrow[r], mr);
            const float al = __expf(mrow[r] - mn);
            const float p0 = __expf(x0 - mn);
            const float p1 = __expf(x1 - mn);
            float rs = p0 + p1;
#pragma unroll
            for (int off = 1; off < 16; off <<= 1)
                rs += __shfl_xor(rs, off, 32);
            lrow[r] = lrow[r] * al + rs;
            mrow[r] = mn;
            alpha[r] = al;
            const int row = ((lane >> 4) << 3) + r;     // D-layout row
            Pst[wave][row][m]      = p0;
            Pst[wave][row][16 + m] = p1;
        }
#pragma unroll
        for (int f = 0; f < 8; ++f)
#pragma unroll
            for (int r = 0; r < 8; ++r) o[f][r] *= alpha[r];

        // D-layout -> A-layout relayout of P (wave-private LDS; same-wave
        // DS ops are in-order, but fence explicitly before the re-read)
        asm volatile("s_wait_dscnt 0" ::: "memory");
        v16u pa;
        {
            const float* pr = &Pst[wave][m][0];
            v4f c0 = *(const v4f*)(pr + kb);
            v4f c1 = *(const v4f*)(pr + kb + 4);
            v4f c2 = *(const v4f*)(pr + kb + 16);
            v4f c3 = *(const v4f*)(pr + kb + 20);
#pragma unroll
            for (int i = 0; i < 4; ++i) {
                pa[i]      = f2bf(c0[i]);  pa[4 + i]  = f2bf(c1[i]);
                pa[8 + i]  = f2bf(c2[i]);  pa[12 + i] = f2bf(c3[i]);
            }
        }

        // O += P @ V : 8 output column tiles of 16 dims
#pragma unroll
        for (int f = 0; f < 8; ++f) {
            const int dv = f * 16 + m;
            const unsigned short* vr = Vti + (size_t)dv * 32 + kb;
            v8u lo = *(const v8u*)vr, hi = *(const v8u*)(vr + 16);
            v16u bf;
#pragma unroll
            for (int i = 0; i < 8; ++i) { bf[i] = lo[i]; bf[8 + i] = hi[i]; }
            o[f] = __builtin_amdgcn_wmma_f32_16x16x32_bf16(
                     false, as_bf(pa), false, as_bf(bf), (short)0, o[f], false, false);
        }
        __syncthreads();              // all waves done with buf before rewrite
    }

    // epilogue: O /= l, store f32 (lanes within a half write contiguous cols)
    float inv[8];
#pragma unroll
    for (int r = 0; r < 8; ++r) inv[r] = 1.0f / lrow[r];
    const int mrow0 = ((lane >> 4) << 3);
    const size_t obase = (size_t)grp * SEQ + qrow0;
#pragma unroll
    for (int f = 0; f < 8; ++f) {
        const int col = f * 16 + m;
#pragma unroll
        for (int r = 0; r < 8; ++r)
            Out[(obase + mrow0 + r) * D_V + col] = o[f][r] * inv[r];
    }
}

// ---------------------------------------------------------------------------
extern "C" void kernel_launch(void* const* d_in, const int* in_sizes, int n_in,
                              void* d_out, int out_size, void* d_ws, size_t ws_size,
                              hipStream_t stream) {
    (void)in_sizes; (void)n_in; (void)out_size; (void)ws_size;
    const float* X  = (const float*)d_in[0];
    // d_in[1] = int64 group index: groups are contiguous SEQ-sized runs; unused
    const float* Wq = (const float*)d_in[2];
    const float* bq = (const float*)d_in[3];
    const float* Wk = (const float*)d_in[4];
    const float* bk = (const float*)d_in[5];
    const float* Wv = (const float*)d_in[6];
    const float* bv = (const float*)d_in[7];

    unsigned short* Qw = (unsigned short*)d_ws;                 // [N,64] bf16
    unsigned short* Kw = Qw + (size_t)NTOK * D_QK;              // [N,64] bf16
    unsigned short* Vt = Kw + (size_t)NTOK * D_QK;              // [G,128,512] bf16

    qkv_proj_kernel<<<512, 256, 0, stream>>>(X, Wq, bq, Wk, bk, Wv, bv, Qw, Kw, Vt);
    flash_attn_kernel<<<dim3(GROUPS, SEQ / 64), 128, 0, stream>>>(
        Qw, Kw, Vt, (float*)d_out);
}
